// SubConv2_85839216378225
// MI455X (gfx1250) — matrix operand order repaired and checked by reading, not compiled, not run on hardware
//
#include <hip/hip_runtime.h>

// ---------------- types ----------------
typedef __attribute__((ext_vector_type(16))) __bf16          v16bf;
typedef __attribute__((ext_vector_type(8)))  float           v8f;
typedef __attribute__((ext_vector_type(16))) unsigned short  v16u;
typedef __attribute__((ext_vector_type(8)))  unsigned short  v8u;

union BF16Frag { v16u u; v16bf b; };

// ---------------- problem constants ----------------
#define N1 1728            // 64*3*3*3
#define N2 36864           // 64*64*3*3
#define N3 4194304         // 256*16384
#define N4 65536           // 256*256
#define N5 2560            // 10*256
#define NTOT 4300992
#define JRANK 2150496u     // int((1-0.5)*NTOT) weights zeroed; keep ranks >= JRANK

// ---------------- workspace layout (bytes) ----------------
#define OFF_HIST1 (0ull)
#define OFF_HIST2 (256ull*1024)
#define OFF_SEL   (512ull*1024)          // [0]=hi bin, [1]=remaining rank, [2]=threshold bits
#define OFF_W1M   (513ull*1024)          // fp32 [64][27]
#define OFF_W2MB  (521ull*1024)          // bf16 [64][9][64]  (co, r=kh*3+kw, ci)
#define OFF_F2M   (600ull*1024)          // fp32 [256][256]
#define OFF_F3M   (872ull*1024)          // fp32 [10][256]
#define OFF_F1MB  (1024ull*1024)         // bf16 [256][16384]
#define OFF_ACT1  (16ull*1024*1024)      // bf16 NHWC [512][32][32][64]
#define OFF_ACT2  (80ull*1024*1024)      // bf16 NHWC [512][32][32][64]
#define OFF_POOL  (144ull*1024*1024)     // bf16 [512][16384] (feature = c*256+y*16+x)
#define OFF_OUT1  (160ull*1024*1024)     // fp32 [512][256]
#define OFF_OUT2  (161ull*1024*1024)     // fp32 [512][256]

// ---------------- helpers ----------------
__device__ __forceinline__ unsigned short f2bf(float f) {
  unsigned u = __float_as_uint(f);
  u += 0x7FFFu + ((u >> 16) & 1u);      // round-to-nearest-even
  return (unsigned short)(u >> 16);
}
__device__ __forceinline__ float bf2f(unsigned short h) {
  return __uint_as_float(((unsigned)h) << 16);
}

__device__ __forceinline__ float score_at(int i,
    const float* s1, const float* s2, const float* s3,
    const float* s4, const float* s5) {
  if (i < N1) return s1[i];
  i -= N1;
  if (i < N2) return s2[i];
  i -= N2;
  if (i < N3) return s3[i];
  i -= N3;
  if (i < N4) return s4[i];
  return s5[i - N4];
}

// ---------------- kernel 0: zero histograms + sel ----------------
__global__ void zero_kernel(unsigned* __restrict__ p, int n) {
  int i = blockIdx.x * blockDim.x + threadIdx.x;
  if (i < n) p[i] = 0u;
}

// ---------------- kernel 1: histogram of top-16 bits of |score| ----------------
__global__ void hist1_kernel(const float* __restrict__ s1, const float* __restrict__ s2,
                             const float* __restrict__ s3, const float* __restrict__ s4,
                             const float* __restrict__ s5, unsigned* __restrict__ hist) {
  int stride = gridDim.x * blockDim.x;
  for (int i = blockIdx.x * blockDim.x + threadIdx.x; i < NTOT; i += stride) {
    unsigned u = (__float_as_uint(score_at(i, s1, s2, s3, s4, s5)) & 0x7FFFFFFFu) >> 16;
    atomicAdd(&hist[u], 1u);
  }
}

// ---------------- kernel 2: find high bucket holding rank JRANK ----------------
__global__ void select1_kernel(const unsigned* __restrict__ hist, unsigned* __restrict__ sel) {
  __shared__ unsigned partial[1024];
  __shared__ unsigned prefix[1024];
  const unsigned t = threadIdx.x;
  unsigned sum = 0;
  for (int i = 0; i < 64; ++i) sum += hist[t * 64 + i];
  partial[t] = sum;
  __syncthreads();
  if (t == 0) {
    unsigned run = 0;
    for (int i = 0; i < 1024; ++i) { prefix[i] = run; run += partial[i]; }
  }
  __syncthreads();
  const unsigned target = JRANK;
  if (prefix[t] <= target && target < prefix[t] + partial[t]) {
    unsigned cum = prefix[t];
    for (int i = 0; i < 64; ++i) {
      unsigned h = hist[t * 64 + i];
      if (target < cum + h) { sel[0] = t * 64 + i; sel[1] = target - cum; break; }
      cum += h;
    }
  }
}

// ---------------- kernel 3: low-16-bit histogram within selected bucket ----------------
__global__ void hist2_kernel(const float* __restrict__ s1, const float* __restrict__ s2,
                             const float* __restrict__ s3, const float* __restrict__ s4,
                             const float* __restrict__ s5, unsigned* __restrict__ hist,
                             const unsigned* __restrict__ sel) {
  const unsigned sb = sel[0];
  int stride = gridDim.x * blockDim.x;
  for (int i = blockIdx.x * blockDim.x + threadIdx.x; i < NTOT; i += stride) {
    unsigned bits = __float_as_uint(score_at(i, s1, s2, s3, s4, s5)) & 0x7FFFFFFFu;
    if ((bits >> 16) == sb) atomicAdd(&hist[bits & 0xFFFFu], 1u);
  }
}

// ---------------- kernel 4: recover exact 32-bit threshold ----------------
__global__ void select2_kernel(const unsigned* __restrict__ hist, unsigned* __restrict__ sel) {
  __shared__ unsigned partial[1024];
  __shared__ unsigned prefix[1024];
  const unsigned t = threadIdx.x;
  const unsigned target = sel[1];
  unsigned sum = 0;
  for (int i = 0; i < 64; ++i) sum += hist[t * 64 + i];
  partial[t] = sum;
  __syncthreads();
  if (t == 0) {
    unsigned run = 0;
    for (int i = 0; i < 1024; ++i) { prefix[i] = run; run += partial[i]; }
  }
  __syncthreads();
  if (prefix[t] <= target && target < prefix[t] + partial[t]) {
    unsigned cum = prefix[t];
    for (int i = 0; i < 64; ++i) {
      unsigned h = hist[t * 64 + i];
      if (target < cum + h) { sel[2] = (sel[0] << 16) | (t * 64 + i); break; }
      cum += h;
    }
  }
}

// ---------------- kernel 5: mask + reorder + precision-convert all weights ----------------
__global__ void prep_kernel(const float* __restrict__ w1, const float* __restrict__ w2,
                            const float* __restrict__ f1, const float* __restrict__ f2,
                            const float* __restrict__ f3,
                            const float* __restrict__ s1, const float* __restrict__ s2,
                            const float* __restrict__ s3, const float* __restrict__ s4,
                            const float* __restrict__ s5,
                            const unsigned* __restrict__ sel,
                            float* __restrict__ w1m, unsigned short* __restrict__ w2mb,
                            unsigned short* __restrict__ f1mb,
                            float* __restrict__ f2m, float* __restrict__ f3m) {
  const float t = __uint_as_float(sel[2]);
  int stride = gridDim.x * blockDim.x;
  for (int i = blockIdx.x * blockDim.x + threadIdx.x; i < NTOT; i += stride) {
    int j = i;
    if (j < N1) {
      w1m[j] = (fabsf(s1[j]) >= t) ? w1[j] : 0.0f;
      continue;
    }
    j -= N1;
    if (j < N2) {
      int co = j / 576, rem = j % 576;
      int ci = rem / 9, r = rem % 9;          // r = kh*3+kw
      float v = (fabsf(s2[j]) >= t) ? w2[j] : 0.0f;
      w2mb[co * 576 + r * 64 + ci] = f2bf(v); // layout [co][r][ci]
      continue;
    }
    j -= N2;
    if (j < N3) {
      float v = (fabsf(s3[j]) >= t) ? f1[j] : 0.0f;
      f1mb[j] = f2bf(v);
      continue;
    }
    j -= N3;
    if (j < N4) {
      f2m[j] = (fabsf(s4[j]) >= t) ? f2[j] : 0.0f;
      continue;
    }
    j -= N4;
    f3m[j] = (fabsf(s5[j]) >= t) ? f3[j] : 0.0f;
  }
}

// ---------------- kernel 6: conv1 (3->64, 3x3, pad1) + relu, NCHW fp32 -> NHWC bf16 ----------------
__global__ void conv1_kernel(const float* __restrict__ x, const float* __restrict__ w1m,
                             unsigned short* __restrict__ act1) {
  __shared__ float wl[N1];
  for (int i = threadIdx.x; i < N1; i += 256) wl[i] = w1m[i];
  __syncthreads();
  int idx = blockIdx.x * 256 + threadIdx.x;           // (pixel, co-group of 16)
  if (idx >= 512 * 1024 * 4) return;
  const int cg = idx & 3;
  const int p  = idx >> 2;                            // b*1024 + y*32 + x
  const int xx = p & 31, yy = (p >> 5) & 31, b = p >> 10;
  const int co0 = cg * 16;
  float acc[16];
#pragma unroll
  for (int c = 0; c < 16; ++c) acc[c] = 0.0f;
#pragma unroll
  for (int ci = 0; ci < 3; ++ci) {
#pragma unroll
    for (int dy = -1; dy <= 1; ++dy) {
#pragma unroll
      for (int dx = -1; dx <= 1; ++dx) {
        int iy = yy + dy, ix = xx + dx;
        if ((unsigned)iy < 32u && (unsigned)ix < 32u) {
          float xv = x[((b * 3 + ci) * 32 + iy) * 32 + ix];
          int off = ci * 9 + (dy + 1) * 3 + (dx + 1);
#pragma unroll
          for (int c = 0; c < 16; ++c)
            acc[c] = fmaf(xv, wl[(co0 + c) * 27 + off], acc[c]);
        }
      }
    }
  }
  v8u o0, o1;
#pragma unroll
  for (int c = 0; c < 8; ++c) {
    o0[c] = f2bf(fmaxf(acc[c], 0.0f));
    o1[c] = f2bf(fmaxf(acc[c + 8], 0.0f));
  }
  unsigned short* dst = act1 + (size_t)p * 64 + co0;  // NHWC
  *(v8u*)dst = o0;
  *(v8u*)(dst + 8) = o1;
}

// ---------------- kernel 7: conv2 implicit GEMM, async LDS staging + WMMA bf16 ----------------
// Block: 256 threads = 8 waves, owns 128 consecutive NHWC pixels (4 x-rows of one image).
// Stage rows y0-1..y0+4 (6 rows x 32px x 64ch bf16 = 24KB) into LDS with
// GLOBAL_LOAD_ASYNC_TO_LDS_B128, then each wave computes 16 pixels x all 64 co
// (4 accumulators) so each LDS B-fragment feeds 4 WMMAs.
__global__ void conv2_wmma(const unsigned short* __restrict__ act1,
                           const unsigned short* __restrict__ w2mb,
                           unsigned short* __restrict__ act2) {
  __shared__ unsigned short smem[6 * 2048];           // 6 rows x (32*64)
  const int tid = threadIdx.x;
  const int p0  = blockIdx.x * 128;                   // first pixel of block
  const int b   = p0 >> 10;
  const int yy0 = (p0 >> 5) & 31;                     // first row (multiple of 4)

  // ---- async stage: thread t copies 16B chunk t of each valid row ----
  const unsigned lds_base = (unsigned)(unsigned long long)(uintptr_t)smem;
#pragma unroll
  for (int r = 0; r < 6; ++r) {
    int iy = yy0 - 1 + r;
    if ((unsigned)iy < 32u) {
      unsigned lds_off = lds_base + (unsigned)(r * 4096 + tid * 16);
      const unsigned short* g = act1 + ((size_t)(b * 32 + iy) * 2048 + tid * 8);
      asm volatile("global_load_async_to_lds_b128 %0, %1, off"
                   :: "v"(lds_off), "v"(g) : "memory");
    }
  }
  asm volatile("s_wait_asynccnt 0x0" ::: "memory");
  __syncthreads();

  const int lane  = tid & 31;
  const int wv    = tid >> 5;                         // 0..7
  const int lmod  = lane & 15;
  const int khalf = lane >> 4;
  const int lpix  = wv * 16 + lmod;                   // 0..127 local pixel
  const int xx    = lpix & 31;
  const int prow  = lpix >> 5;                        // 0..3
  const int yy    = yy0 + prow;

  v8f zero = {};
  v8f acc[4];
#pragma unroll
  for (int mt = 0; mt < 4; ++mt) acc[mt] = zero;

  for (int kb = 0; kb < 576; kb += 32) {
    const int r   = kb >> 6;                          // kh*3+kw
    const int dy  = r / 3 - 1, dx = r % 3 - 1;
    const int ci0 = (kb & 63) + khalf * 16;
    // B fragment from LDS: N=lane%16 pixel, K=khalf*16+s -> 16 contiguous channels
    BF16Frag B;
    {
      int iy = yy + dy, ix = xx + dx;
      if ((unsigned)iy < 32u && (unsigned)ix < 32u) {
        int lrow = prow + dy + 1;                     // 0..5
        B.u = *(const v16u*)(smem + (lrow * 2048 + ix * 64 + ci0));
      } else {
#pragma unroll
        for (int s = 0; s < 16; ++s) B.u[s] = 0;      // zero padding
      }
    }
#pragma unroll
    for (int mt = 0; mt < 4; ++mt) {
      const unsigned short* arow = w2mb + (mt * 16 + lmod) * 576;
      BF16Frag A;
      v8u lo = *(const v8u*)(arow + kb + khalf * 16);
      v8u hi = *(const v8u*)(arow + kb + khalf * 16 + 8);
#pragma unroll
      for (int s = 0; s < 8; ++s) { A.u[s] = lo[s]; A.u[s + 8] = hi[s]; }
      acc[mt] = __builtin_amdgcn_wmma_f32_16x16x32_bf16(false, A.b, false, B.b,
                                                        (short)0, acc[mt], false, false);
    }
  }
  // store all 64 co for this pixel: 4 x 16B contiguous NHWC stores (relu deferred to pool)
  unsigned short* dst = act2 + (((size_t)(b * 32 + yy) * 32 + xx) * 64 + khalf * 8);
#pragma unroll
  for (int mt = 0; mt < 4; ++mt) {
    v8u o;
#pragma unroll
    for (int i = 0; i < 8; ++i) o[i] = f2bf(acc[mt][i]);
    *(v8u*)(dst + mt * 16) = o;
  }
}

// ---------------- kernel 8: 2x2 maxpool + relu, NHWC bf16 -> [b][c*256+y*16+x] bf16 ----------------
__global__ void pool_kernel(const unsigned short* __restrict__ act2,
                            unsigned short* __restrict__ pooled) {
  int idx = blockIdx.x * blockDim.x + threadIdx.x;    // 512*16384
  if (idx >= 512 * 16384) return;
  const int k = idx & 16383, b = idx >> 14;
  const int c = k >> 8, rem = k & 255, y = rem >> 4, x = rem & 15;
  const unsigned short* base = act2 + (size_t)b * 65536 + c;
  float m0 = bf2f(base[((2 * y) * 32 + 2 * x) * 64]);
  float m1 = bf2f(base[((2 * y) * 32 + 2 * x + 1) * 64]);
  float m2 = bf2f(base[((2 * y + 1) * 32 + 2 * x) * 64]);
  float m3 = bf2f(base[((2 * y + 1) * 32 + 2 * x + 1) * 64]);
  float m = fmaxf(fmaxf(m0, m1), fmaxf(m2, m3));
  m = fmaxf(m, 0.0f);                                  // relu(maxpool) == maxpool(relu)
  pooled[idx] = f2bf(m);
}

// ---------------- kernel 9: fc1 via WMMA bf16, 2x2 register blocking, relu, fp32 out ----------------
// C[256 feat, 512 batch]; each wave computes a 32x32 output block (4 accumulators).
__global__ void fc1_wmma(const unsigned short* __restrict__ f1mb,
                         const unsigned short* __restrict__ pooled,
                         float* __restrict__ out1) {
  const int wave = (blockIdx.x * blockDim.x + threadIdx.x) >> 5;  // 128 waves
  const int lane = threadIdx.x & 31;
  const int mg = wave & 7;                    // 8 groups of 32 features
  const int ng = wave >> 3;                   // 16 groups of 32 batches
  const int lmod  = lane & 15;
  const int khalf = lane >> 4;
  const unsigned short* arow0 = f1mb + (size_t)(mg * 32 + lmod) * 16384;
  const unsigned short* arow1 = arow0 + 16 * 16384;
  const unsigned short* brow0 = pooled + (size_t)(ng * 32 + lmod) * 16384;
  const unsigned short* brow1 = brow0 + 16 * 16384;
  v8f zero = {};
  v8f acc[4];                                 // [mi*2+ni]
#pragma unroll
  for (int i = 0; i < 4; ++i) acc[i] = zero;
  for (int kb = 0; kb < 16384; kb += 32) {
    __builtin_prefetch(arow0 + kb + 512, 0, 0);
    __builtin_prefetch(brow0 + kb + 512, 0, 0);
    BF16Frag A0, A1, B0, B1;
    {
      v8u lo0 = *(const v8u*)(arow0 + kb + khalf * 16);
      v8u hi0 = *(const v8u*)(arow0 + kb + khalf * 16 + 8);
      v8u lo1 = *(const v8u*)(arow1 + kb + khalf * 16);
      v8u hi1 = *(const v8u*)(arow1 + kb + khalf * 16 + 8);
#pragma unroll
      for (int s = 0; s < 8; ++s) {
        A0.u[s] = lo0[s]; A0.u[s + 8] = hi0[s];
        A1.u[s] = lo1[s]; A1.u[s + 8] = hi1[s];
      }
    }
    B0.u = *(const v16u*)(brow0 + kb + khalf * 16);
    B1.u = *(const v16u*)(brow1 + kb + khalf * 16);
    acc[0] = __builtin_amdgcn_wmma_f32_16x16x32_bf16(false, A0.b, false, B0.b,
                                                     (short)0, acc[0], false, false);
    acc[1] = __builtin_amdgcn_wmma_f32_16x16x32_bf16(false, A0.b, false, B1.b,
                                                     (short)0, acc[1], false, false);
    acc[2] = __builtin_amdgcn_wmma_f32_16x16x32_bf16(false, A1.b, false, B0.b,
                                                     (short)0, acc[2], false, false);
    acc[3] = __builtin_amdgcn_wmma_f32_16x16x32_bf16(false, A1.b, false, B1.b,
                                                     (short)0, acc[3], false, false);
  }
#pragma unroll
  for (int mi = 0; mi < 2; ++mi) {
#pragma unroll
    for (int ni = 0; ni < 2; ++ni) {
      v8f r;
      v8f a = acc[mi * 2 + ni];
#pragma unroll
      for (int i = 0; i < 8; ++i) r[i] = fmaxf(a[i], 0.0f);
      const int nB = ng * 32 + ni * 16 + lmod;             // batch
      const int mB = mg * 32 + mi * 16 + khalf * 8;        // feature base
      *(v8f*)(out1 + nB * 256 + mB) = r;
    }
  }
}

// ---------------- kernel 10: fc2 (256->256) fp32 + relu ----------------
__global__ void fc2_kernel(const float* __restrict__ out1, const float* __restrict__ f2m,
                           float* __restrict__ out2) {
  int idx = blockIdx.x * blockDim.x + threadIdx.x;    // 512*256
  if (idx >= 512 * 256) return;
  const int j = idx & 255, b = idx >> 8;
  const float* a = out1 + b * 256;
  const float* w = f2m + j * 256;
  float acc = 0.0f;
  for (int k = 0; k < 256; ++k) acc = fmaf(a[k], w[k], acc);
  out2[b * 256 + j] = fmaxf(acc, 0.0f);
}

// ---------------- kernel 11: fc3 (256->10) fp32 ----------------
__global__ void fc3_kernel(const float* __restrict__ out2, const float* __restrict__ f3m,
                           float* __restrict__ out) {
  int idx = blockIdx.x * blockDim.x + threadIdx.x;    // 512*10
  if (idx >= 512 * 10) return;
  const int j = idx % 10, b = idx / 10;
  const float* a = out2 + b * 256;
  const float* w = f3m + j * 256;
  float acc = 0.0f;
  for (int k = 0; k < 256; ++k) acc = fmaf(a[k], w[k], acc);
  out[b * 10 + j] = acc;
}

// ---------------- host-side launch ----------------
extern "C" void kernel_launch(void* const* d_in, const int* in_sizes, int n_in,
                              void* d_out, int out_size, void* d_ws, size_t ws_size,
                              hipStream_t stream) {
  (void)in_sizes; (void)n_in; (void)out_size; (void)ws_size;
  const float* x  = (const float*)d_in[0];
  const float* w1 = (const float*)d_in[1];
  const float* w2 = (const float*)d_in[2];
  const float* f1 = (const float*)d_in[3];
  const float* f2 = (const float*)d_in[4];
  const float* f3 = (const float*)d_in[5];
  const float* s1 = (const float*)d_in[6];
  const float* s2 = (const float*)d_in[7];
  const float* s3 = (const float*)d_in[8];
  const float* s4 = (const float*)d_in[9];
  const float* s5 = (const float*)d_in[10];

  char* ws = (char*)d_ws;
  unsigned*       hist1  = (unsigned*)(ws + OFF_HIST1);
  unsigned*       hist2  = (unsigned*)(ws + OFF_HIST2);
  unsigned*       sel    = (unsigned*)(ws + OFF_SEL);
  float*          w1m    = (float*)(ws + OFF_W1M);
  unsigned short* w2mb   = (unsigned short*)(ws + OFF_W2MB);
  unsigned short* f1mb   = (unsigned short*)(ws + OFF_F1MB);
  float*          f2m    = (float*)(ws + OFF_F2M);
  float*          f3m    = (float*)(ws + OFF_F3M);
  unsigned short* act1   = (unsigned short*)(ws + OFF_ACT1);
  unsigned short* act2   = (unsigned short*)(ws + OFF_ACT2);
  unsigned short* pooled = (unsigned short*)(ws + OFF_POOL);
  float*          out1   = (float*)(ws + OFF_OUT1);
  float*          out2   = (float*)(ws + OFF_OUT2);

  // 1) exact global median threshold over |scores| (2-pass radix select)
  zero_kernel<<<513, 256, 0, stream>>>((unsigned*)ws, 131328);   // hist1+hist2+sel
  hist1_kernel<<<4096, 256, 0, stream>>>(s1, s2, s3, s4, s5, hist1);
  select1_kernel<<<1, 1024, 0, stream>>>(hist1, sel);
  hist2_kernel<<<4096, 256, 0, stream>>>(s1, s2, s3, s4, s5, hist2, sel);
  select2_kernel<<<1, 1024, 0, stream>>>(hist2, sel);

  // 2) masked weights (+ bf16 conversion, + w2 reorder to [co][r][ci])
  prep_kernel<<<4096, 256, 0, stream>>>(w1, w2, f1, f2, f3,
                                        s1, s2, s3, s4, s5, sel,
                                        w1m, w2mb, f1mb, f2m, f3m);

  // 3) network forward
  conv1_kernel<<<8192, 256, 0, stream>>>(x, w1m, act1);          // 2M threads x 16 co
  conv2_wmma<<<4096, 256, 0, stream>>>(act1, w2mb, act2);        // 32768 waves x 72 wmma
  pool_kernel<<<32768, 256, 0, stream>>>(act2, pooled);
  fc1_wmma<<<16, 256, 0, stream>>>(f1mb, pooled, out1);          // 128 waves x 2048 wmma
  fc2_kernel<<<512, 256, 0, stream>>>(out1, f2m, out2);
  fc3_kernel<<<20, 256, 0, stream>>>(out2, f3m, (float*)d_out);
}
